// Block_76244259438867
// MI455X (gfx1250) — matrix operand order repaired and checked by reading, not compiled, not run on hardware
//
#include <hip/hip_runtime.h>
#include <hip/hip_bf16.h>
#include <math.h>

typedef _Float16 half_t;
typedef __attribute__((ext_vector_type(16))) _Float16 v16h;
typedef __attribute__((ext_vector_type(8)))  float    v8f;

#define EPSF 1e-9f

// ---------------------------------------------------------------------------
// CDNA5 async global->LDS copy (ISA 15.18.3, tracked by ASYNCcnt).
// VDST = VGPR holding LDS byte address, VADDR = 64-bit global address.
// Async loads complete IN ORDER, so s_wait_asynccnt N drains all but the
// newest N outstanding copies -> double-buffered pipelining.
// ---------------------------------------------------------------------------
__device__ __forceinline__ void async_ld_b128(void* lds_dst, const void* gsrc) {
    unsigned ldsa = (unsigned)(size_t)lds_dst;           // LDS addr = addr[31:0]
    unsigned long long ga = (unsigned long long)(size_t)gsrc;
    asm volatile("global_load_async_to_lds_b128 %0, %1, off"
                 :: "v"(ldsa), "v"(ga) : "memory");
}
__device__ __forceinline__ void wait_async0() {
    asm volatile("s_wait_asynccnt 0x0" ::: "memory");
}
__device__ __forceinline__ void wait_async_le8() {
    asm volatile("s_wait_asynccnt 0x8" ::: "memory");
}
__device__ __forceinline__ void wait_async_le16() {
    asm volatile("s_wait_asynccnt 0x10" ::: "memory");
}

// ---------------------------------------------------------------------------
// WMMA fragment loaders (contiguous k = 16*(lane/16) + i order, wave32)
// ---------------------------------------------------------------------------
__device__ __forceinline__ v16h load_frag_a(const half_t* base, int stride) {
    int l  = threadIdx.x & 31;
    int m  = l & 15;
    int ko = (l >> 4) << 4;
    const half_t* p = base + (size_t)m * stride;
    v16h a;
#pragma unroll
    for (int h = 0; h < 16; ++h) {
        int k = h + ko;
        a[h] = p[k];
    }
    return a;
}

// B[kc][n] where memory holds rows indexed by n: elem = base[n*stride + kc]
__device__ __forceinline__ v16h load_frag_b_nk(const half_t* base, int stride) {
    int l  = threadIdx.x & 31;
    int n  = l & 15;
    int kb = (l >> 4) << 4;
    const half_t* p = base + (size_t)n * stride + kb;
    v16h b;
#pragma unroll
    for (int h = 0; h < 16; ++h) b[h] = p[h];
    return b;
}

// B[kc][n] where memory holds rows indexed by kc: elem = base[kc*stride + n]
__device__ __forceinline__ v16h load_frag_b_kn(const half_t* base, int stride) {
    int l  = threadIdx.x & 31;
    int n  = l & 15;
    int kb = (l >> 4) << 4;
    v16h b;
#pragma unroll
    for (int h = 0; h < 16; ++h) b[h] = base[(size_t)(kb + h) * stride + n];
    return b;
}

__device__ __forceinline__ v8f wmma_f16(v16h a, v16h b, v8f c) {
    return __builtin_amdgcn_wmma_f32_16x16x32_f16(false, a, false, b, (short)0, c,
                                                  false, false);
}

// ---------------------------------------------------------------------------
// fp32 -> fp16 conversion
// ---------------------------------------------------------------------------
__global__ void f32_to_f16_kernel(const float* __restrict__ in,
                                  half_t* __restrict__ out, int n) {
    int i = blockIdx.x * blockDim.x + threadIdx.x;
    if (i < n) out[i] = (half_t)in[i];
}

// ---------------------------------------------------------------------------
// LayerNorm over C=1024, writes f16. One block (256 threads) per row.
// ---------------------------------------------------------------------------
__device__ __forceinline__ float block_reduce_sum(float v, float* sh) {
#pragma unroll
    for (int o = 16; o; o >>= 1) v += __shfl_xor(v, o, 32);
    int w = threadIdx.x >> 5, l = threadIdx.x & 31;
    __syncthreads();
    if (l == 0) sh[w] = v;
    __syncthreads();
    float r = 0.f;
#pragma unroll
    for (int i = 0; i < 8; ++i) r += sh[i];
    return r;
}

__global__ void ln_kernel(const float* __restrict__ x, const float* __restrict__ g,
                          half_t* __restrict__ out) {
    __shared__ float sh[8];
    const int row = blockIdx.x, tid = threadIdx.x;
    const float4 v = ((const float4*)(x + (size_t)row * 1024))[tid];
    float s = v.x + v.y + v.z + v.w;
    float q = v.x * v.x + v.y * v.y + v.z * v.z + v.w * v.w;
    s = block_reduce_sum(s, sh);
    q = block_reduce_sum(q, sh);
    const float mean = s * (1.f / 1024.f);
    const float var  = q * (1.f / 1024.f) - mean * mean;
    const float rs   = rsqrtf(var + 1e-5f);
    const float4 gg  = ((const float4*)g)[tid];
    half_t* op = out + (size_t)row * 1024 + tid * 4;
    op[0] = (half_t)((v.x - mean) * rs * gg.x);
    op[1] = (half_t)((v.y - mean) * rs * gg.y);
    op[2] = (half_t)((v.z - mean) * rs * gg.z);
    op[3] = (half_t)((v.w - mean) * rs * gg.w);
}

// ---------------------------------------------------------------------------
// WMMA GEMM:  C[M,N] = A[M,K] (f16) * W[N,K]^T (f16), fp32 accumulate.
// Block: 256 threads (8 waves), 128x128 tile, KT=64, double-buffered LDS.
// Next K-slab staged via async-to-LDS WHILE current slab is computed;
// s_wait_asynccnt 8 (in-order completion) gates only the previous slab.
// ---------------------------------------------------------------------------
__global__ void gemm_wmma_kernel(const half_t* __restrict__ A,
                                 const half_t* __restrict__ W,
                                 void* __restrict__ Cout,
                                 int M, int N, int K, int outF16, int gelu) {
    __shared__ half_t sA[2][128 * 64];
    __shared__ half_t sW[2][128 * 64];
    const int tid = threadIdx.x;
    const int wid = tid >> 5;
    const int wm  = wid & 3;     // 32-row strip
    const int wn  = wid >> 2;    // 64-col strip
    const int m0  = blockIdx.y * 128;
    const int n0  = blockIdx.x * 128;

    v8f acc[2][4];
#pragma unroll
    for (int im = 0; im < 2; ++im)
#pragma unroll
        for (int jn = 0; jn < 4; ++jn) acc[im][jn] = v8f{};

    auto stage = [&](int buf, int k0) {
#pragma unroll
        for (int i = 0; i < 4; ++i) {
            int idx = (i * 256 + tid) * 8;   // 0..8191 halves
            int r = idx >> 6, c = idx & 63;
            async_ld_b128(sA[buf] + idx, A + (size_t)(m0 + r) * K + k0 + c);
            async_ld_b128(sW[buf] + idx, W + (size_t)(n0 + r) * K + k0 + c);
        }
    };

    const int nk = K >> 6;
    stage(0, 0);
    for (int ki = 0; ki < nk; ++ki) {
        const int cur = ki & 1;
        if (ki + 1 < nk) {
            stage(cur ^ 1, (ki + 1) << 6);   // prefetch next slab (no wait)
            wait_async_le8();                // drain current slab only
        } else {
            wait_async0();
        }
        __syncthreads();
#pragma unroll
        for (int ks = 0; ks < 64; ks += 32) {
            v16h a[2];
#pragma unroll
            for (int im = 0; im < 2; ++im)
                a[im] = load_frag_a(sA[cur] + (size_t)(wm * 32 + im * 16) * 64 + ks, 64);
#pragma unroll
            for (int jn = 0; jn < 4; ++jn) {
                v16h b = load_frag_b_nk(sW[cur] + (size_t)(wn * 64 + jn * 16) * 64 + ks, 64);
#pragma unroll
                for (int im = 0; im < 2; ++im)
                    acc[im][jn] = wmma_f16(a[im], b, acc[im][jn]);
            }
        }
        __syncthreads();
    }

    const int l = tid & 31;
    const int n = l & 15;
    const int mg = (l >> 4) << 3;
#pragma unroll
    for (int im = 0; im < 2; ++im) {
#pragma unroll
        for (int jn = 0; jn < 4; ++jn) {
#pragma unroll
            for (int r = 0; r < 8; ++r) {
                float v = acc[im][jn][r];
                if (gelu) v = 0.5f * v * (1.f + erff(v * 0.70710678f));
                size_t row = (size_t)(m0 + wm * 32 + im * 16 + mg + r);
                size_t col = (size_t)(n0 + wn * 64 + jn * 16 + n);
                if (outF16)
                    ((half_t*)Cout)[row * N + col] = (half_t)v;
                else
                    ((float*)Cout)[row * N + col] = v;
            }
        }
    }
}

// ---------------------------------------------------------------------------
// Flash-style causal attention, one wave per (b, h, 16-query tile).
// Key tiles of 32; scores via 4 WMMAs, PV via 4 WMMAs.
// K/V tiles double-buffered via async-to-LDS; next tile prefetched while the
// current one is computed (s_wait_asynccnt 16 gates only the previous tile).
// ---------------------------------------------------------------------------
__global__ void attn_kernel(const half_t* __restrict__ qkv, half_t* __restrict__ y) {
    // per-wave: q 16x64 | k[2] 32x64 | v[2] 32x64 | p 16x32  = 9728 halves
    __shared__ half_t smem[4 * 9728];
    const int tid = threadIdx.x, wid = tid >> 5, lane = tid & 31;
    half_t* sq = smem + wid * 9728;
    half_t* skb[2] = { sq + 1024, sq + 1024 + 2048 };
    half_t* svb[2] = { sq + 1024 + 4096, sq + 1024 + 4096 + 2048 };
    half_t* sp = sq + 1024 + 8192;

    const int wave = blockIdx.x * 4 + wid;
    const int bh = wave >> 7;          // 0..31
    const int qt = wave & 127;         // 0..127
    const int b = bh >> 4, h = bh & 15;
    const int qBase = qt * 16;
    const size_t ld = 3072;

    auto stage_kv = [&](int it2, int buf) {
        const int kt2 = it2 * 32;
        const half_t* kg = qkv + ((size_t)(b * 2048 + kt2)) * ld + 1024 + h * 64;
        const half_t* krow = kg + (size_t)lane * ld;
        const half_t* vrow = krow + 1024;
        half_t* kdst = skb[buf] + lane * 64;
        half_t* vdst = svb[buf] + lane * 64;
#pragma unroll
        for (int i = 0; i < 8; ++i) {
            async_ld_b128(kdst + i * 8, krow + i * 8);
            async_ld_b128(vdst + i * 8, vrow + i * 8);
        }
    };

    // stage Q tile (16x64) + first K/V tile
    {
        const half_t* qg = qkv + ((size_t)(b * 2048 + qBase)) * ld + h * 64;
        int r = lane >> 1, cb = (lane & 1) * 32;
        const half_t* qrow = qg + (size_t)r * ld + cb;
        half_t* qdst = sq + r * 64 + cb;
#pragma unroll
        for (int i = 0; i < 4; ++i) async_ld_b128(qdst + i * 8, qrow + i * 8);
    }
    stage_kv(0, 0);

    float rowm[8], rowl[8];
#pragma unroll
    for (int r = 0; r < 8; ++r) { rowm[r] = -1e30f; rowl[r] = 0.f; }
    v8f o0 = {}, o1 = {}, o2 = {}, o3 = {};

    const int mg = (lane >> 4) << 3;
    const int nl = lane & 15;
    const int nIter = ((qBase + 15) >> 5) + 1;

    for (int it = 0; it < nIter; ++it) {
        const int cur = it & 1;
        if (it + 1 < nIter) {
            stage_kv(it + 1, cur ^ 1);   // prefetch next K/V tile
            wait_async_le16();           // drain Q + current tile (in-order)
        } else {
            wait_async0();
        }
        const half_t* sk = skb[cur];
        const half_t* sv = svb[cur];
        const int kt = it * 32;

        // S = Q * K^T  (16x32 as two 16x16 accs), contraction over d=64
        v8f s0 = {}, s1 = {};
#pragma unroll
        for (int ks = 0; ks < 64; ks += 32) {
            v16h a = load_frag_a(sq + ks, 64);
            v16h b0 = load_frag_b_nk(sk + 0 * 64 + ks, 64);
            s0 = wmma_f16(a, b0, s0);
            v16h b1 = load_frag_b_nk(sk + 16 * 64 + ks, 64);
            s1 = wmma_f16(a, b1, s1);
        }

        // online softmax (C layout: row m = mg+r, cols nl and nl+16)
        const int n0g = kt + nl, n1g = n0g + 16;
#pragma unroll
        for (int r = 0; r < 8; ++r) {
            const int qi = qBase + mg + r;
            float e0 = s0[r] * 0.125f;
            float e1 = s1[r] * 0.125f;
            if (n0g > qi) e0 = -1e30f;
            if (n1g > qi) e1 = -1e30f;
            float tm = fmaxf(e0, e1);
#pragma unroll
            for (int o = 8; o; o >>= 1) tm = fmaxf(tm, __shfl_xor(tm, o, 16));
            const float nm  = fmaxf(rowm[r], tm);
            const float scl = __expf(rowm[r] - nm);
            const float p0  = __expf(e0 - nm);
            const float p1  = __expf(e1 - nm);
            float ts = p0 + p1;
#pragma unroll
            for (int o = 8; o; o >>= 1) ts += __shfl_xor(ts, o, 16);
            rowl[r] = rowl[r] * scl + ts;
            rowm[r] = nm;
            o0[r] *= scl; o1[r] *= scl; o2[r] *= scl; o3[r] *= scl;
            const int m = mg + r;
            sp[m * 32 + nl]      = (half_t)p0;
            sp[m * 32 + 16 + nl] = (half_t)p1;
        }
        asm volatile("s_wait_dscnt 0x0" ::: "memory");

        // O += P * V   (contraction over 32 keys)
        v16h ap = load_frag_a(sp, 32);
        o0 = wmma_f16(ap, load_frag_b_kn(sv + 0,  64), o0);
        o1 = wmma_f16(ap, load_frag_b_kn(sv + 16, 64), o1);
        o2 = wmma_f16(ap, load_frag_b_kn(sv + 32, 64), o2);
        o3 = wmma_f16(ap, load_frag_b_kn(sv + 48, 64), o3);
    }

    // normalize + store f16
#pragma unroll
    for (int r = 0; r < 8; ++r) {
        const float inv = 1.f / (rowl[r] + 1e-20f);
        size_t row = (size_t)b * 2048 + qBase + mg + r;
        half_t* yr = y + row * 1024 + h * 64;
        yr[0 * 16 + nl] = (half_t)(o0[r] * inv);
        yr[1 * 16 + nl] = (half_t)(o1[r] * inv);
        yr[2 * 16 + nl] = (half_t)(o2[r] * inv);
        yr[3 * 16 + nl] = (half_t)(o3[r] * inv);
    }
}

// ---------------------------------------------------------------------------
// Hyperbolic residual: out = mobius_add(x, coeff(v)*v/|v|, clamp(c)) per head.
// ---------------------------------------------------------------------------
__global__ void expmap_kernel(const float* __restrict__ x, const float* __restrict__ v,
                              const float* __restrict__ c, float* __restrict__ out) {
    const int row = blockIdx.x, tid = threadIdx.x;
    const float4 xv = ((const float4*)(x + (size_t)row * 1024))[tid];
    const float4 vv4 = ((const float4*)(v + (size_t)row * 1024))[tid];
    float xx = xv.x * xv.x + xv.y * xv.y + xv.z * xv.z + xv.w * xv.w;
    float vv = vv4.x * vv4.x + vv4.y * vv4.y + vv4.z * vv4.z + vv4.w * vv4.w;
    float xvip = xv.x * vv4.x + xv.y * vv4.y + xv.z * vv4.z + xv.w * vv4.w;
#pragma unroll
    for (int o = 8; o; o >>= 1) {
        xx += __shfl_xor(xx, o, 16);
        vv += __shfl_xor(vv, o, 16);
        xvip += __shfl_xor(xvip, o, 16);
    }
    const float cc = fminf(fmaxf(c[tid >> 4], 1e-4f), 1.f);
    const float sf = 2.f / (1.f + cc * xx + EPSF);
    const float vn = sqrtf(vv + EPSF);
    const float targ = fabsf(cc * sf * vv * 0.5f);
    const float coeff = (1.f / (sqrtf(fabsf(cc) + EPSF) + EPSF)) *
                        tanhf(sqrtf(targ + EPSF));
    const float s = coeff / (vn + EPSF);     // second = s * v
    const float yn = s * s * vv;
    const float ip = s * xvip;
    const float c1 = 1.f + 2.f * cc * ip + cc * yn;    // multiplies x
    const float c2 = (1.f - cc * xx) * s;              // multiplies v
    const float den = 1.f + 2.f * cc * ip + cc * cc * xx * yn + EPSF;
    const float inv = 1.f / den;
    float4 o;
    o.x = (c1 * xv.x + c2 * vv4.x) * inv;
    o.y = (c1 * xv.y + c2 * vv4.y) * inv;
    o.z = (c1 * xv.z + c2 * vv4.z) * inv;
    o.w = (c1 * xv.w + c2 * vv4.w) * inv;
    ((float4*)(out + (size_t)row * 1024))[tid] = o;
}

// ---------------------------------------------------------------------------
// Orchestration
// ---------------------------------------------------------------------------
extern "C" void kernel_launch(void* const* d_in, const int* in_sizes, int n_in,
                              void* d_out, int out_size, void* d_ws, size_t ws_size,
                              hipStream_t stream) {
    (void)in_sizes; (void)n_in; (void)out_size; (void)ws_size;
    const float* x      = (const float*)d_in[0];
    const float* ln1_g  = (const float*)d_in[1];
    const float* w_qkv  = (const float*)d_in[2];
    const float* w_proj = (const float*)d_in[3];
    const float* c_attn = (const float*)d_in[4];
    const float* ln2_g  = (const float*)d_in[5];
    const float* w_fc   = (const float*)d_in[6];
    const float* w_mlp  = (const float*)d_in[7];
    const float* c_mlp  = (const float*)d_in[8];
    float* out = (float*)d_out;

    char* ws = (char*)d_ws;
    const size_t MB = (size_t)1 << 20;
    half_t* qkv16   = (half_t*)(ws);              // 24 MB, dead after attention
    half_t* ln16    = (half_t*)(ws + 24 * MB);    // 8 MB (ln1), dead after qkv gemm
    half_t* y16     = (half_t*)(ws + 24 * MB);    // alias: written by attention
    half_t* wqkv16  = (half_t*)(ws + 32 * MB);    // 6 MB
    half_t* wproj16 = (half_t*)(ws + 38 * MB);    // 2 MB
    float*  aproj   = (float*) (ws + 40 * MB);    // 16 MB, dead after expmap1
    float*  hmlp    = (float*) (ws + 40 * MB);    // alias
    float*  x1      = (float*) (ws + 56 * MB);    // 16 MB
    half_t* ln2o    = (half_t*)(ws + 72 * MB);    // 8 MB
    half_t* wfc16   = (half_t*)(ws + 80 * MB);    // 8 MB
    half_t* wmlp16  = (half_t*)(ws + 88 * MB);    // 8 MB
    half_t* h1      = (half_t*)(ws);              // alias: 32 MB (qkv16+y16 dead)

    const int M = 4096, C = 1024;

    // weight conversions
    f32_to_f16_kernel<<<(3072 * 1024 + 255) / 256, 256, 0, stream>>>(w_qkv, wqkv16, 3072 * 1024);
    f32_to_f16_kernel<<<(1024 * 1024 + 255) / 256, 256, 0, stream>>>(w_proj, wproj16, 1024 * 1024);
    f32_to_f16_kernel<<<(4096 * 1024 + 255) / 256, 256, 0, stream>>>(w_fc, wfc16, 4096 * 1024);
    f32_to_f16_kernel<<<(1024 * 4096 + 255) / 256, 256, 0, stream>>>(w_mlp, wmlp16, 1024 * 4096);

    // attention half
    ln_kernel<<<M, 256, 0, stream>>>(x, ln1_g, ln16);
    gemm_wmma_kernel<<<dim3(3072 / 128, M / 128), 256, 0, stream>>>(
        ln16, wqkv16, qkv16, M, 3072, C, /*outF16=*/1, /*gelu=*/0);
    attn_kernel<<<1024, 128, 0, stream>>>(qkv16, y16);
    gemm_wmma_kernel<<<dim3(1024 / 128, M / 128), 256, 0, stream>>>(
        y16, wproj16, aproj, M, C, C, /*outF16=*/0, /*gelu=*/0);
    expmap_kernel<<<M, 256, 0, stream>>>(x, aproj, c_attn, x1);

    // MLP half
    ln_kernel<<<M, 256, 0, stream>>>(x1, ln2_g, ln2o);
    gemm_wmma_kernel<<<dim3(4096 / 128, M / 128), 256, 0, stream>>>(
        ln2o, wfc16, h1, M, 4096, C, /*outF16=*/1, /*gelu=*/1);
    gemm_wmma_kernel<<<dim3(1024 / 128, M / 128), 256, 0, stream>>>(
        h1, wmlp16, hmlp, M, C, 4096, /*outF16=*/0, /*gelu=*/0);
    expmap_kernel<<<M, 256, 0, stream>>>(x1, hmlp, c_mlp, out);
}
